// MambaExperts_40853728920337
// MI455X (gfx1250) — matrix operand order repaired and checked by reading, not compiled, not run on hardware
//
#include <hip/hip_runtime.h>

// ---------------------------------------------------------------------------
// Problem constants (from reference)
// ---------------------------------------------------------------------------
constexpr int E_   = 8;
constexpr int B_   = 2;
constexpr int L_   = 2048;
constexpr int DIN  = 1024;
constexpr int DM   = 512;
constexpr int DI   = 1024;
constexpr int DS   = 64;
constexpr int DR   = 32;
constexpr int KC   = 4;
constexpr int NL   = 2;
constexpr int NC   = 2;
constexpr int ROWS = B_ * L_;          // 4096 rows per expert
constexpr float EPS = 1e-5f;

// gfx1250 async global->LDS path (ASYNCcnt), with safe fallback
#if defined(__gfx1250__) && __has_builtin(__builtin_amdgcn_global_load_async_to_lds_b128) && __has_builtin(__builtin_amdgcn_s_wait_asynccnt)
#define ASYNC_LDS 1
#else
#define ASYNC_LDS 0
#endif

#if ASYNC_LDS
// builtin signature (from hipcc diagnostics):
//   param0 = vector_size(16) int, AS(1) ("__device__"), non-const
//   param1 = vector_size(16) int, AS(3) ("__shared__")
typedef int v4i_ __attribute__((vector_size(4 * sizeof(int))));
typedef __attribute__((address_space(1))) v4i_* async_gptr_t;
typedef __attribute__((address_space(3))) v4i_* async_lptr_t;
#endif

// ---------------------------------------------------------------------------
// CDNA5 WMMA types
// ---------------------------------------------------------------------------
typedef __attribute__((ext_vector_type(16))) __bf16 v16bf;
typedef __attribute__((ext_vector_type(8)))  float  v8f;

union FragBF {
    v16bf v;
    uint4 q[2];
};

__device__ __forceinline__ unsigned short f2bf(float f) {
    unsigned int u = __float_as_uint(f);
    unsigned int r = u + 0x7FFFu + ((u >> 16) & 1u);   // round-to-nearest-even
    return (unsigned short)(r >> 16);
}

__device__ __forceinline__ float block_reduce_sum(float v, float* red) {
    int tid = threadIdx.x;
    red[tid] = v;
    __syncthreads();
    for (int off = 128; off > 0; off >>= 1) {
        if (tid < off) red[tid] += red[tid + off];
        __syncthreads();
    }
    float r = red[0];
    __syncthreads();
    return r;
}

__device__ __forceinline__ float epilogue(float c, int act) {
    if (act == 1) return fmaxf(c, 0.0f);
    if (act == 2) return (c > 20.0f) ? c : __logf(1.0f + __expf(c));
    return c;
}

// ---------------------------------------------------------------------------
// Elementwise f32 -> bf16 conversion
// ---------------------------------------------------------------------------
__global__ void __launch_bounds__(256)
cvt_kernel(const float* __restrict__ src, unsigned short* __restrict__ dst, int n) {
    int i = blockIdx.x * 256 + threadIdx.x;
    if (i < n) dst[i] = f2bf(src[i]);
}

// strided column-slice convert (dt = dbl[:, 0:32] -> bf16)
__global__ void __launch_bounds__(256)
slicecvt_kernel(const float* __restrict__ src, int srcld, int coloff,
                unsigned short* __restrict__ dst, int width, int total) {
    int i = blockIdx.x * 256 + threadIdx.x;
    if (i >= total) return;
    int row = i / width, c = i % width;
    dst[i] = f2bf(src[(size_t)row * srcld + coloff + c]);
}

// ---------------------------------------------------------------------------
// Block-tiled WMMA bf16 GEMM: C[M,N] = act(A[M,K] @ W[N,K]^T + bias + resid)
// 256 threads = 8 waves; block computes a 128x128 tile; K-panels of 32 staged
// in LDS (async global->LDS on gfx1250, ds_store fallback). Each wave owns a
// 16-row strip and runs 8 v_wmma_f32_16x16x32_bf16 per K-step, reusing its A
// fragment across all 8 N-tiles (block-level reuse: ~64 FLOP/byte from L2).
// Requires M%128==0, N%128==0, K%32==0.
// ---------------------------------------------------------------------------
__global__ void __launch_bounds__(256)
wmma_gemm_tiled_kernel(const unsigned short* __restrict__ A, int lda,
                       const unsigned short* __restrict__ W, int ldw,
                       const float* __restrict__ bias,
                       const float* __restrict__ resid,
                       float* __restrict__ C, int ldc,
                       int K, int act) {
    __shared__ uint4 AshQ[512];                 // 128 rows x 32 bf16 = 8KB
    __shared__ uint4 BshQ[512];                 // 128 rows x 32 bf16 = 8KB
    unsigned short* Ash = (unsigned short*)AshQ;
    unsigned short* Bsh = (unsigned short*)BshQ;

    const int tmBase = blockIdx.y * 128;
    const int tnBase = blockIdx.x * 128;
    const int lane = threadIdx.x;               // 0..31 (wave32)
    const int w    = threadIdx.y;               // 0..7
    const int tidFlat = w * 32 + lane;
    const int laneHi = lane >> 4;
    const int laneLo = lane & 15;

    v8f zero = {};
    v8f acc[8];
#pragma unroll
    for (int t = 0; t < 8; ++t) acc[t] = zero;

    for (int kk = 0; kk < K; kk += 32) {
        // cooperative panel copy: 512 x 16B chunks each for A and B
#pragma unroll
        for (int i = tidFlat; i < 512; i += 256) {
            int row = i >> 2, qc = i & 3;
            const uint4* ga = reinterpret_cast<const uint4*>(
                A + (size_t)(tmBase + row) * lda + kk + qc * 8);
            const uint4* gb = reinterpret_cast<const uint4*>(
                W + (size_t)(tnBase + row) * ldw + kk + qc * 8);
#if ASYNC_LDS
            __builtin_amdgcn_global_load_async_to_lds_b128(
                (async_gptr_t)ga, (async_lptr_t)&AshQ[i], 0, 0);
            __builtin_amdgcn_global_load_async_to_lds_b128(
                (async_gptr_t)gb, (async_lptr_t)&BshQ[i], 0, 0);
#else
            AshQ[i] = *ga;
            BshQ[i] = *gb;
#endif
        }
#if ASYNC_LDS
        __builtin_amdgcn_s_wait_asynccnt(0);
#endif
        __syncthreads();

        // A fragment for this wave's 16-row strip (16x32 bf16, ISA 7.12.2 layout)
        FragBF fa;
        const unsigned short* arow = Ash + (w * 16 + laneLo) * 32;
        fa.q[0] = *reinterpret_cast<const uint4*>(arow + 8 * laneHi);
        fa.q[1] = *reinterpret_cast<const uint4*>(arow + 8 * laneHi + 16);

#pragma unroll
        for (int t = 0; t < 8; ++t) {
            FragBF fb;                           // B 32x16: same K order as A (8*laneHi, +16)
            const unsigned short* brow = Bsh + (t * 16 + laneLo) * 32;
            fb.q[0] = *reinterpret_cast<const uint4*>(brow + 8 * laneHi);
            fb.q[1] = *reinterpret_cast<const uint4*>(brow + 8 * laneHi + 16);
            acc[t] = __builtin_amdgcn_wmma_f32_16x16x32_bf16(
                false, fa.v, false, fb.v, (short)0, acc[t], false, false);
        }
        __syncthreads();
    }

#pragma unroll
    for (int t = 0; t < 8; ++t) {
        int n = tnBase + t * 16 + laneLo;
        float bn = bias ? bias[n] : 0.0f;
#pragma unroll
        for (int r = 0; r < 8; ++r) {
            int row = tmBase + w * 16 + r + 8 * laneHi;   // C/D: M = r + 8*laneHi
            size_t idx = (size_t)row * ldc + n;
            float c = acc[t][r] + bn;
            if (resid) c += resid[idx];
            C[idx] = epilogue(c, act);
        }
    }
}

// ---------------------------------------------------------------------------
// Fallback WMMA GEMM for N not divisible by 128 (x-proj, N=160):
// one wave per 16x16 tile, fragments streamed from L2.
// ---------------------------------------------------------------------------
__global__ void __launch_bounds__(256)
wmma_gemm_small_kernel(const unsigned short* __restrict__ A, int lda,
                       const unsigned short* __restrict__ W, int ldw,
                       const float* __restrict__ bias,
                       const float* __restrict__ resid,
                       float* __restrict__ C, int ldc,
                       int M, int N, int K, int act) {
    int lane   = threadIdx.x;
    int wave   = threadIdx.y;
    int tilesN = N >> 4;
    int tile   = blockIdx.x * 8 + wave;
    int total  = (M >> 4) * tilesN;
    if (tile >= total) return;                 // wave-uniform: EXEC stays all-1s

    int tm = (tile / tilesN) << 4;
    int tn = (tile % tilesN) << 4;
    int laneHi = lane >> 4;
    int laneLo = lane & 15;
    int m = tm + laneLo;
    int n = tn + laneLo;

    const unsigned short* Arow = A + (size_t)m * lda;
    const unsigned short* Wrow = W + (size_t)n * ldw;

    v8f acc = {};
    for (int kk = 0; kk < K; kk += 32) {
        FragBF fa, fb;
        int ka = kk + 8 * laneHi;
        fa.q[0] = *reinterpret_cast<const uint4*>(Arow + ka);
        fa.q[1] = *reinterpret_cast<const uint4*>(Arow + ka + 16);
        int kb = kk + 8 * laneHi;
        fb.q[0] = *reinterpret_cast<const uint4*>(Wrow + kb);
        fb.q[1] = *reinterpret_cast<const uint4*>(Wrow + kb + 16);
        acc = __builtin_amdgcn_wmma_f32_16x16x32_bf16(
            false, fa.v, false, fb.v, (short)0, acc, false, false);
    }

    float bn = bias ? bias[n] : 0.0f;
#pragma unroll
    for (int r = 0; r < 8; ++r) {
        int row = tm + r + 8 * laneHi;
        size_t idx = (size_t)row * ldc + n;
        float c = acc[r] + bn;
        if (resid) c += resid[idx];
        C[idx] = epilogue(c, act);
    }
}

// ---------------------------------------------------------------------------
// RMSNorm over DM=512; one 256-thread block per row; bf16 out
// ---------------------------------------------------------------------------
__global__ void __launch_bounds__(256)
rmsnorm_kernel(const float* __restrict__ h, const float* __restrict__ rw,
               unsigned short* __restrict__ out) {
    __shared__ float red[256];
    size_t row = blockIdx.x;
    int tid = threadIdx.x;
    const float* hr = h + row * DM;
    float v0 = hr[tid], v1 = hr[tid + 256];
    float ss = block_reduce_sum(v0 * v0 + v1 * v1, red);
    float rinv = rsqrtf(ss * (1.0f / DM) + EPS);
    out[row * DM + tid]       = f2bf(rw[tid] * v0 * rinv);
    out[row * DM + tid + 256] = f2bf(rw[tid + 256] * v1 * rinv);
}

// ---------------------------------------------------------------------------
// Causal depthwise conv (K=4) + SiLU; xc lives in columns [0,DI) of xz
// ---------------------------------------------------------------------------
__global__ void __launch_bounds__(256)
conv_silu_kernel(const float* __restrict__ xz, const float* __restrict__ cw,
                 const float* __restrict__ cb, float* __restrict__ u,
                 unsigned short* __restrict__ u16) {
    int idx = blockIdx.x * 256 + threadIdx.x;   // over ROWS*DI
    if (idx >= ROWS * DI) return;
    int d = idx % DI;
    int r = idx / DI;
    int l = r % L_;
    int b = r / L_;
    float acc = cb[d];
#pragma unroll
    for (int k = 0; k < KC; ++k) {
        int l2 = l + k - (KC - 1);
        if (l2 >= 0)
            acc += cw[d * KC + k] * xz[((size_t)(b * L_ + l2)) * (2 * DI) + d];
    }
    float s = acc / (1.0f + __expf(-acc));       // silu
    u[idx] = s;
    u16[idx] = f2bf(s);
}

// ---------------------------------------------------------------------------
// Selective scan. One block = 256 channels of one batch element.
// 64 states + per-channel A row held in VGPRs; B_t/C_t broadcast via LDS.
// Epilogue fuses  y = scan + u*D, gating by silu(z), bf16 store for out-proj.
// ---------------------------------------------------------------------------
__global__ void __launch_bounds__(256)
scan_kernel(const float* __restrict__ delta, const float* __restrict__ u,
            const float* __restrict__ dbl, const float* __restrict__ xz,
            const float* __restrict__ Alog, const float* __restrict__ Dp,
            unsigned short* __restrict__ y16) {
    __shared__ float Bs[DS];
    __shared__ float Cs[DS];
    int blocksPerB = DI / 256;
    int b    = blockIdx.x / blocksPerB;
    int dblk = blockIdx.x % blocksPerB;
    int d    = dblk * 256 + threadIdx.x;

    float hs[DS];
    float As[DS];
#pragma unroll
    for (int s = 0; s < DS; ++s) {
        hs[s] = 0.0f;
        As[s] = -__expf(Alog[(size_t)d * DS + s]);
    }
    float Dd = Dp[d];

    for (int t = 0; t < L_; ++t) {
        size_t row = (size_t)b * L_ + t;
        __syncthreads();
        if (threadIdx.x < DS)
            Bs[threadIdx.x] = dbl[row * 160 + DR + threadIdx.x];
        else if (threadIdx.x < 2 * DS)
            Cs[threadIdx.x - DS] = dbl[row * 160 + DR + DS + (threadIdx.x - DS)];
        if (threadIdx.x == 0 && t + 1 < L_)
            __builtin_prefetch(dbl + (row + 1) * 160 + DR, 0, 1); // global_prefetch_b8
        __syncthreads();

        float dt = delta[row * DI + d];
        float ut = u[row * DI + d];
        float du = dt * ut;
        float y = 0.0f;
#pragma unroll
        for (int s = 0; s < DS; ++s) {
            float dec = __expf(dt * As[s]);
            hs[s] = dec * hs[s] + du * Bs[s];
            y += hs[s] * Cs[s];
        }
        y += ut * Dd;
        float z = xz[row * (2 * DI) + DI + d];
        float g = z / (1.0f + __expf(-z));       // silu(z)
        y16[row * DI + d] = f2bf(y * g);
    }
}

// ---------------------------------------------------------------------------
// Final LayerNorm over DM=512 (one block per row)
// ---------------------------------------------------------------------------
__global__ void __launch_bounds__(256)
ln_kernel(const float* __restrict__ h, const float* __restrict__ g,
          const float* __restrict__ bta, float* __restrict__ out) {
    __shared__ float red[256];
    size_t row = blockIdx.x;
    int tid = threadIdx.x;
    const float* hr = h + row * DM;
    float v0 = hr[tid], v1 = hr[tid + 256];
    float mu = block_reduce_sum(v0 + v1, red) * (1.0f / DM);
    float d0 = v0 - mu, d1 = v1 - mu;
    float var = block_reduce_sum(d0 * d0 + d1 * d1, red) * (1.0f / DM);
    float rstd = rsqrtf(var + EPS);
    out[row * DM + tid]       = d0 * rstd * g[tid] + bta[tid];
    out[row * DM + tid + 256] = d1 * rstd * g[tid + 256] + bta[tid + 256];
}

// mean-pool over L into feat[e][b][dm]
__global__ void __launch_bounds__(256)
colsum_kernel(const float* __restrict__ lnbuf, float* __restrict__ featOut) {
    int idx = blockIdx.x * 256 + threadIdx.x;   // B_*DM = 1024
    if (idx >= B_ * DM) return;
    int b = idx >> 9, dm = idx & 511;
    float s = 0.0f;
    for (int l = 0; l < L_; ++l)
        s += lnbuf[((size_t)(b * L_ + l)) * DM + dm];
    featOut[(size_t)b * DM + dm] = s * (1.0f / L_);
}

// ---------------------------------------------------------------------------
// Head: transpose features, classifier logits, MoE mean + logits.
// Output layout: features[8192] | logits[32] | moe_features[1024] | moe_logits[4]
// ---------------------------------------------------------------------------
__global__ void __launch_bounds__(256)
head_kernel(const float* __restrict__ feat, const float* __restrict__ wcls,
            const float* __restrict__ bcls, float* __restrict__ out) {
    __shared__ float moes[B_ * DM];
    int tid = threadIdx.x;

    for (int i = tid; i < B_ * E_ * DM; i += 256) {
        int dm = i % DM;
        int e = (i / DM) % E_;
        int b = i / (DM * E_);
        out[i] = feat[((size_t)e * B_ + b) * DM + dm];
    }
    for (int i = tid; i < B_ * DM; i += 256) {
        int dm = i % DM, b = i / DM;
        float s = 0.0f;
        for (int e = 0; e < E_; ++e) s += feat[((size_t)e * B_ + b) * DM + dm];
        s *= (1.0f / E_);
        moes[i] = s;
        out[B_ * E_ * DM + B_ * E_ * NC + i] = s;
    }
    __syncthreads();
    for (int i = tid; i < B_ * E_ * NC; i += 256) {
        int c = i % NC;
        int e = (i / NC) % E_;
        int b = i / (NC * E_);
        float s = bcls[c];
        for (int dm = 0; dm < DM; ++dm)
            s += feat[((size_t)e * B_ + b) * DM + dm] * wcls[c * DM + dm];
        out[B_ * E_ * DM + i] = s;
    }
    for (int i = tid; i < B_ * NC; i += 256) {
        int c = i % NC, b = i / NC;
        float s = bcls[c];
        for (int dm = 0; dm < DM; ++dm)
            s += moes[b * DM + dm] * wcls[c * DM + dm];
        out[B_ * E_ * DM + B_ * E_ * NC + B_ * DM + i] = s;
    }
}

// ---------------------------------------------------------------------------
// Host orchestration
// ---------------------------------------------------------------------------
static inline void launch_gemm(const unsigned short* A, int lda,
                               const unsigned short* W, int ldw,
                               const float* bias, const float* resid,
                               float* C, int ldc, int M, int N, int K, int act,
                               hipStream_t s) {
    if ((M % 128 == 0) && (N % 128 == 0)) {
        dim3 grid(N / 128, M / 128), blk(32, 8);
        wmma_gemm_tiled_kernel<<<grid, blk, 0, s>>>(A, lda, W, ldw, bias, resid,
                                                    C, ldc, K, act);
    } else {
        int tiles = (M / 16) * (N / 16);
        wmma_gemm_small_kernel<<<(tiles + 7) / 8, dim3(32, 8), 0, s>>>(
            A, lda, W, ldw, bias, resid, C, ldc, M, N, K, act);
    }
}

static inline void launch_cvt(const float* src, unsigned short* dst, int n,
                              hipStream_t s) {
    cvt_kernel<<<(n + 255) / 256, 256, 0, s>>>(src, dst, n);
}

extern "C" void kernel_launch(void* const* d_in, const int* in_sizes, int n_in,
                              void* d_out, int out_size, void* d_ws, size_t ws_size,
                              hipStream_t stream) {
    (void)in_sizes; (void)n_in; (void)out_size; (void)ws_size;

    const float* x        = (const float*)d_in[0];
    const float* w_in     = (const float*)d_in[1];
    const float* b_in     = (const float*)d_in[2];
    const float* in_proj  = (const float*)d_in[3];
    const float* conv_w   = (const float*)d_in[4];
    const float* conv_b   = (const float*)d_in[5];
    const float* x_proj   = (const float*)d_in[6];
    const float* dt_w     = (const float*)d_in[7];
    const float* dt_b     = (const float*)d_in[8];
    const float* A_log    = (const float*)d_in[9];
    const float* D_ssm    = (const float*)d_in[10];
    const float* out_proj = (const float*)d_in[11];
    const float* rms_w    = (const float*)d_in[12];
    const float* ln_g     = (const float*)d_in[13];
    const float* ln_b     = (const float*)d_in[14];
    const float* w_cls    = (const float*)d_in[15];
    const float* b_cls    = (const float*)d_in[16];
    float* out = (float*)d_out;

    char* ws = (char*)d_ws;
    size_t o = 0;
    auto alloc = [&](size_t bytes) -> char* {
        char* p = ws + o;
        o = (o + bytes + 255) & ~(size_t)255;
        return p;
    };
    float*          feat   = (float*)alloc((size_t)E_ * B_ * DM * 4);
    unsigned short* x16    = (unsigned short*)alloc((size_t)ROWS * DIN * 2);
    unsigned short* win16  = (unsigned short*)alloc((size_t)DM * DIN * 2);
    unsigned short* inw16  = (unsigned short*)alloc((size_t)2 * DI * DM * 2);
    unsigned short* xpw16  = (unsigned short*)alloc((size_t)(DR + 2 * DS) * DI * 2);
    unsigned short* dtw16  = (unsigned short*)alloc((size_t)DI * DR * 2);
    unsigned short* ow16   = (unsigned short*)alloc((size_t)DM * DI * 2);
    float*          h      = (float*)alloc((size_t)ROWS * DM * 4);
    unsigned short* hn16   = (unsigned short*)alloc((size_t)ROWS * DM * 2);
    float*          xz     = (float*)alloc((size_t)ROWS * 2 * DI * 4);
    float*          u      = (float*)alloc((size_t)ROWS * DI * 4);
    unsigned short* u16    = (unsigned short*)alloc((size_t)ROWS * DI * 2);
    float*          dbl    = (float*)alloc((size_t)ROWS * 160 * 4);
    unsigned short* dt16   = (unsigned short*)alloc((size_t)ROWS * DR * 2);
    float*          delta  = (float*)alloc((size_t)ROWS * DI * 4);
    unsigned short* y16    = (unsigned short*)alloc((size_t)ROWS * DI * 2);
    float*          lnbuf  = (float*)alloc((size_t)ROWS * DM * 4);

    for (int e = 0; e < E_; ++e) {
        // input projection: h = relu(x @ w_in^T + b_in)
        launch_cvt(x + (size_t)e * ROWS * DIN, x16, ROWS * DIN, stream);
        launch_cvt(w_in + (size_t)e * DM * DIN, win16, DM * DIN, stream);
        launch_gemm(x16, DIN, win16, DIN, b_in + (size_t)e * DM, nullptr,
                    h, DM, ROWS, DM, DIN, /*relu*/1, stream);

        for (int l = 0; l < NL; ++l) {
            size_t el = (size_t)e * NL + l;
            launch_cvt(in_proj + el * 2 * DI * DM, inw16, 2 * DI * DM, stream);
            launch_cvt(x_proj + el * (DR + 2 * DS) * DI, xpw16, (DR + 2 * DS) * DI, stream);
            launch_cvt(dt_w + el * DI * DR, dtw16, DI * DR, stream);
            launch_cvt(out_proj + el * DM * DI, ow16, DM * DI, stream);

            // hn = rmsnorm(h) * rms_w  (bf16)
            rmsnorm_kernel<<<ROWS, 256, 0, stream>>>(h, rms_w + el * DM, hn16);
            // xz = hn @ in_proj^T
            launch_gemm(hn16, DM, inw16, DM, nullptr, nullptr,
                        xz, 2 * DI, ROWS, 2 * DI, DM, 0, stream);
            // u = silu(conv(xc) + conv_b)
            conv_silu_kernel<<<(ROWS * DI) / 256, 256, 0, stream>>>(
                xz, conv_w + el * DI * KC, conv_b + el * DI, u, u16);
            // dbl = u @ x_proj^T  (dt|B|C)   N=160 -> small-kernel path
            launch_gemm(u16, DI, xpw16, DI, nullptr, nullptr,
                        dbl, 160, ROWS, 160, DI, 0, stream);
            // dt16 = bf16(dbl[:, :32])
            slicecvt_kernel<<<(ROWS * DR) / 256, 256, 0, stream>>>(
                dbl, 160, 0, dt16, DR, ROWS * DR);
            // delta = softplus(dt @ dt_w^T + dt_b)
            launch_gemm(dt16, DR, dtw16, DR, dt_b + el * DI, nullptr,
                        delta, DI, ROWS, DI, DR, /*softplus*/2, stream);
            // selective scan + u*D + silu(z) gating -> y16 (bf16)
            scan_kernel<<<B_ * (DI / 256), 256, 0, stream>>>(
                delta, u, dbl, xz, A_log + el * DI * DS, D_ssm + el * DI, y16);
            // h = h + y16 @ out_proj^T  (residual fused)
            launch_gemm(y16, DI, ow16, DI, nullptr, /*resid*/h,
                        h, DM, ROWS, DM, DI, 0, stream);
        }

        // LayerNorm + mean over L -> feat[e]
        ln_kernel<<<ROWS, 256, 0, stream>>>(h, ln_g, ln_b, lnbuf);
        colsum_kernel<<<(B_ * DM) / 256, 256, 0, stream>>>(
            lnbuf, feat + (size_t)e * B_ * DM);
    }

    head_kernel<<<1, 256, 0, stream>>>(feat, w_cls, b_cls, out);
}
